// PermuteToFrom_34316788695850
// MI455X (gfx1250) — compile-verified, matrix operand order checked by reading, not run on hardware
//
#include <hip/hip_runtime.h>
#include <hip/hip_fp16.h>
#include <math.h>

typedef __attribute__((ext_vector_type(16))) _Float16 v16h;
typedef __attribute__((ext_vector_type(8)))  float    v8f;
typedef __attribute__((ext_vector_type(4)))  float    f4;

// Problem constants
static constexpr int CB   = 4;
static constexpr int CC   = 64;
static constexpr int CH   = 128;
static constexpr int CW   = 128;
static constexpr int KTAP = 9;
static constexpr int KTOT = CC * KTAP;   // 576 ; K index kk = tap*64 + c
static constexpr int NKC  = KTOT / 32;   // 18 chunks; tap = kc>>1 (scalar/chunk)

// Pair-packed fragment tiles: u32 p = (k>>1)*16 + n holds halves (k, k+1).
static constexpr int BTILE_U32 = 4608;   // (576/2)*16 per 16-wide N tile

// ---------------------------------------------------------------------------
__device__ inline v16h load_frag(const unsigned* __restrict__ tu, int mn, int hi, int k0) {
  union { v16h h; unsigned u[8]; } f;
#pragma unroll
  for (int v = 0; v < 8; ++v) {
    int kv = k0 + ((v >> 2) << 4) + (hi << 3) + ((v & 3) << 1);
    f.u[v] = tu[((kv >> 1) << 4) + mn];
  }
  return f.h;
}

__device__ inline unsigned pack2(float v0, float v1) {
  union { _Float16 h[2]; unsigned u; } q;
  q.h[0] = (_Float16)v0; q.h[1] = (_Float16)v1;
  return q.u;
}

// Async global->LDS copy (ASYNCcnt path, ISA 15.18.3 op 98)
__device__ inline void async_copy_b128(unsigned lds_byte, const void* gptr) {
  asm volatile("global_load_async_to_lds_b128 %0, %1, off"
               :: "v"(lds_byte), "v"((unsigned long long)(uintptr_t)gptr)
               : "memory");
}
__device__ inline void wait_async0() {
  asm volatile("s_wait_asynccnt 0x0" ::: "memory");
}
// Wave-private LDS tiles only need a compiler/scheduler fence (DS in-order per wave)
__device__ inline void wavebar() {
  __builtin_amdgcn_wave_barrier();
  asm volatile("" ::: "memory");
}

// ---------------------------------------------------------------------------
// Kernel 1: pack weights f32 -> f16 fragment layout, K order kk = tap*64 + c.
// B1: [576 x 32] (18 offset_w | 9 mod_w | 5 zero)   B2: [576 x 64] reg_w
// ---------------------------------------------------------------------------
__global__ void pack_weights(const float* __restrict__ offw,
                             const float* __restrict__ modw,
                             const float* __restrict__ regw,
                             _Float16* __restrict__ b1,
                             _Float16* __restrict__ b2) {
  int id = blockIdx.x * blockDim.x + threadIdx.x;
  if (id >= KTOT * 96) return;
  int n   = id % 96;
  int kk  = id / 96;
  int tap = kk >> 6;
  int c   = kk & 63;
  if (n < 32) {
    float v = 0.0f;
    if (n < 18)      v = offw[(n * CC + c) * KTAP + tap];
    else if (n < 27) v = modw[((n - 18) * CC + c) * KTAP + tap];
    int p = (n >> 4) * BTILE_U32 + ((kk >> 1) << 4) + (n & 15);
    b1[p * 2 + (kk & 1)] = (_Float16)v;
  } else {
    int o = n - 32;
    float v = regw[(o * CC + c) * KTAP + tap];
    int p = (o >> 4) * BTILE_U32 + ((kk >> 1) << 4) + (o & 15);
    b2[p * 2 + (kk & 1)] = (_Float16)v;
  }
}

// ---------------------------------------------------------------------------
// Kernel 2: offset/mask conv as fused im2col WMMA GEMM. One WG per (b,h) row.
// ---------------------------------------------------------------------------
__global__ void __launch_bounds__(256)
conv_offset_mask(const float* __restrict__ x,
                 const unsigned* __restrict__ b1,
                 const float* __restrict__ offb,
                 const float* __restrict__ modb,
                 float* __restrict__ offset_out,
                 float* __restrict__ mask_out) {
  extern __shared__ unsigned smem[];
  unsigned*  Bt = smem;                                   // 2*4608 u32 (36 KB)
  _Float16*  At = (_Float16*)(smem + 2 * BTILE_U32);      // 8 * 512 halves (8 KB)

  const int tid  = threadIdx.x;
  const int lane = tid & 31;
  const int wave = tid >> 5;
  const int b    = blockIdx.x >> 7;
  const int h    = blockIdx.x & 127;

  {
    unsigned lb = (unsigned)(uintptr_t)Bt;
    for (int i = tid; i < (2 * BTILE_U32) / 4; i += 256)   // 2304 b128, 9/thread
      async_copy_b128(lb + i * 16, b1 + i * 4);
    wait_async0();
  }
  __syncthreads();

  const int w0 = wave * 16;
  const int hi = lane >> 4;
  const int mn = lane & 15;
  const int p  = lane & 15;     // channel-pair index
  const int mh = lane >> 4;     // m half
  _Float16* Awh  = At + wave * 512;
  unsigned* Aw32 = (unsigned*)Awh;

  v8f acc0 = {}; v8f acc1 = {};

  for (int kc = 0; kc < NKC; ++kc) {
    const int  k0  = kc * 32;
    const int  tap = kc >> 1;            // scalar per chunk
    const int  c0  = (kc & 1) * 32;
    const int  kx  = tap % 3 - 1;
    const int  y   = h + tap / 3 - 1;
    const bool yv  = (unsigned)y < (unsigned)CH;
    const int  ycl = min(max(y, 0), CH - 1);
    const float* xr0 = x + ((b * CC + c0 + 2 * p) * CH + ycl) * CW;
    const float* xr1 = xr0 + CH * CW;
#pragma unroll
    for (int i = 0; i < 8; ++i) {
      int  m   = mh * 8 + i;
      int  xw  = w0 + m + kx;
      bool ok  = yv && ((unsigned)xw < (unsigned)CW);
      int  xcl = min(max(xw, 0), CW - 1);
      float v0 = xr0[xcl];
      float v1 = xr1[xcl];
      v0 = ok ? v0 : 0.0f;
      v1 = ok ? v1 : 0.0f;
      Aw32[p * 16 + m] = pack2(v0, v1);
    }
    wavebar();
    v16h a  = load_frag(Aw32, mn, hi, 0);
    v16h f0 = load_frag(Bt,             mn, hi, k0);
    v16h f1 = load_frag(Bt + BTILE_U32, mn, hi, k0);
    acc0 = __builtin_amdgcn_wmma_f32_16x16x32_f16(false, a, false, f0, (short)0, acc0, false, false);
    acc1 = __builtin_amdgcn_wmma_f32_16x16x32_f16(false, a, false, f1, (short)0, acc1, false, false);
    wavebar();
  }

  // ---- epilogue: vectorized b128 stores ----
  const int wbase = w0 + hi * 8;
  {
    int n = mn;                                   // offset channels 0..15
    float bias = offb[n];
    float* dst = offset_out + ((b * 18 + n) * CH + h) * CW + wbase;
    f4 s0 = {acc0[0] + bias, acc0[1] + bias, acc0[2] + bias, acc0[3] + bias};
    f4 s1 = {acc0[4] + bias, acc0[5] + bias, acc0[6] + bias, acc0[7] + bias};
    *(f4*)dst = s0; *(f4*)(dst + 4) = s1;
  }
  {
    int n = 16 + mn;
    if (n < 18) {                                 // offset channels 16,17
      float bias = offb[n];
      float* dst = offset_out + ((b * 18 + n) * CH + h) * CW + wbase;
      f4 s0 = {acc1[0] + bias, acc1[1] + bias, acc1[2] + bias, acc1[3] + bias};
      f4 s1 = {acc1[4] + bias, acc1[5] + bias, acc1[6] + bias, acc1[7] + bias};
      *(f4*)dst = s0; *(f4*)(dst + 4) = s1;
    } else if (n < 27) {                          // mask channels 0..8
      float bias = modb[n - 18];
      float* dst = mask_out + ((b * 9 + (n - 18)) * CH + h) * CW + wbase;
      f4 s0, s1;
#pragma unroll
      for (int r = 0; r < 4; ++r) s0[r] = 2.0f / (1.0f + __expf(-(acc1[r]     + bias)));
#pragma unroll
      for (int r = 0; r < 4; ++r) s1[r] = 2.0f / (1.0f + __expf(-(acc1[r + 4] + bias)));
      *(f4*)dst = s0; *(f4*)(dst + 4) = s1;
    }
  }
}

// ---------------------------------------------------------------------------
// Kernel 3: fused deformable-conv GEMM. One WG per (b,h) row.
// ---------------------------------------------------------------------------
__global__ void __launch_bounds__(256)
deform_gemm(const float* __restrict__ x,
            const unsigned* __restrict__ b2,
            const float* __restrict__ offset,
            const float* __restrict__ mask,
            float* __restrict__ out) {
  extern __shared__ unsigned smem[];
  unsigned* Bt    = smem;                           // 4*4608 u32 (72 KB)
  unsigned* Samp  = smem + 4 * BTILE_U32;           // 1152*8 u32 (36 KB)
  _Float16* At    = (_Float16*)(Samp + 1152 * 8);   // 8 * 512 halves (8 KB)
  float*    SampF = (float*)Samp;

  const int tid  = threadIdx.x;
  const int lane = tid & 31;
  const int wave = tid >> 5;
  const int b    = blockIdx.x >> 7;
  const int h    = blockIdx.x & 127;

  {
    unsigned lb = (unsigned)(uintptr_t)Bt;
    for (int i = tid; i < (4 * BTILE_U32) / 4; i += 256)   // 4608 b128, 18/thread
      async_copy_b128(lb + i * 16, b2 + i * 4);
  }

  // ---- per-(pixel,tap) bilinear descriptors: 4 clamped addrs + 4 weights ----
  for (int s = tid; s < CW * KTAP; s += 256) {
    int pw = s / KTAP, tap = s % KTAP;
    float offy = offset[((b * 18 + 2 * tap)     * CH + h) * CW + pw];
    float offx = offset[((b * 18 + 2 * tap + 1) * CH + h) * CW + pw];
    float mval = mask  [((b * 9 + tap)          * CH + h) * CW + pw];
    float py = (float)(h  + tap / 3 - 1) + offy;
    float px = (float)(pw + tap % 3 - 1) + offx;
    float y0f = floorf(py), x0f = floorf(px);
    float wy = py - y0f,    wx = px - x0f;
    int y0 = (int)y0f, x0 = (int)x0f;
    int y1 = y0 + 1,   x1 = x0 + 1;
    float vy0 = (y0 >= 0 && y0 < CH) ? 1.0f : 0.0f;
    float vy1 = (y1 >= 0 && y1 < CH) ? 1.0f : 0.0f;
    float vx0 = (x0 >= 0 && x0 < CW) ? 1.0f : 0.0f;
    float vx1 = (x1 >= 0 && x1 < CW) ? 1.0f : 0.0f;
    int cy0 = min(max(y0, 0), CH - 1), cy1 = min(max(y1, 0), CH - 1);
    int cx0 = min(max(x0, 0), CW - 1), cx1 = min(max(x1, 0), CW - 1);
    unsigned* sp = Samp + s * 8;
    sp[0] = (unsigned)(cy0 * CW + cx0);
    sp[1] = (unsigned)(cy0 * CW + cx1);
    sp[2] = (unsigned)(cy1 * CW + cx0);
    sp[3] = (unsigned)(cy1 * CW + cx1);
    SampF[s * 8 + 4] = (1.0f - wy) * (1.0f - wx) * vy0 * vx0 * mval;
    SampF[s * 8 + 5] = (1.0f - wy) * wx          * vy0 * vx1 * mval;
    SampF[s * 8 + 6] = wy * (1.0f - wx)          * vy1 * vx0 * mval;
    SampF[s * 8 + 7] = wy * wx                   * vy1 * vx1 * mval;
  }
  wait_async0();
  __syncthreads();

  const int w0 = wave * 16;
  const int hi = lane >> 4;
  const int mn = lane & 15;
  const int p  = lane & 15;
  const int mh = lane >> 4;
  _Float16* Awh  = At + wave * 512;
  unsigned* Aw32 = (unsigned*)Awh;

  v8f acc[4] = {};

  for (int kc = 0; kc < NKC; ++kc) {
    const int k0  = kc * 32;
    const int tap = kc >> 1;             // scalar per chunk
    const int c0  = (kc & 1) * 32;
    const float* xp0 = x + (size_t)((b * CC + c0 + 2 * p)) * (CH * CW);
    const float* xp1 = xp0 + CH * CW;
#pragma unroll
    for (int i = 0; i < 8; ++i) {
      int m = mh * 8 + i;
      int s = ((w0 + m) * KTAP + tap) * 8;
      uint4 oo = *(const uint4*)(Samp + s);      // b128 LDS read (uniform per 16-lane half)
      f4    ww = *(const f4*)(SampF + s + 4);
      float v0 = ww[0] * xp0[oo.x] + ww[1] * xp0[oo.y]
               + ww[2] * xp0[oo.z] + ww[3] * xp0[oo.w];
      float v1 = ww[0] * xp1[oo.x] + ww[1] * xp1[oo.y]
               + ww[2] * xp1[oo.z] + ww[3] * xp1[oo.w];
      Aw32[p * 16 + m] = pack2(v0, v1);
    }
    wavebar();
    v16h a = load_frag(Aw32, mn, hi, 0);
#pragma unroll
    for (int nt = 0; nt < 4; ++nt) {
      v16h f = load_frag(Bt + nt * BTILE_U32, mn, hi, k0);
      acc[nt] = __builtin_amdgcn_wmma_f32_16x16x32_f16(false, a, false, f, (short)0, acc[nt], false, false);
    }
    wavebar();
  }

  // ---- epilogue: NCHW, 2x b128 store per accumulator ----
  const size_t rowoff = (size_t)h * CW + w0 + hi * 8;
#pragma unroll
  for (int nt = 0; nt < 4; ++nt) {
    float* dst = out + (((size_t)b * CC + (nt * 16 + mn)) << 14) + rowoff;
    f4 s0 = {acc[nt][0], acc[nt][1], acc[nt][2], acc[nt][3]};
    f4 s1 = {acc[nt][4], acc[nt][5], acc[nt][6], acc[nt][7]};
    *(f4*)dst = s0; *(f4*)(dst + 4) = s1;
  }
}

// ---------------------------------------------------------------------------
extern "C" void kernel_launch(void* const* d_in, const int* in_sizes, int n_in,
                              void* d_out, int out_size, void* d_ws, size_t ws_size,
                              hipStream_t stream) {
  const float* x    = (const float*)d_in[0];
  const float* offw = (const float*)d_in[1];
  const float* offb = (const float*)d_in[2];
  const float* modw = (const float*)d_in[3];
  const float* modb = (const float*)d_in[4];
  const float* regw = (const float*)d_in[5];

  char* ws = (char*)d_ws;
  _Float16* b1         = (_Float16*)ws;                          // 36864 B
  _Float16* b2         = (_Float16*)(ws + 36864);                // 73728 B
  float*    offset_buf = (float*)(ws + 36864 + 73728);           // 4.5 MB
  float*    mask_buf   = (float*)(ws + 36864 + 73728 + 4718592); // 2.25 MB

  pack_weights<<<(KTOT * 96 + 255) / 256, 256, 0, stream>>>(offw, modw, regw, b1, b2);

  conv_offset_mask<<<CB * CH, 256, (2 * BTILE_U32) * 4 + 8 * 1024, stream>>>(
      x, (const unsigned*)b1, offb, modb, offset_buf, mask_buf);

  deform_gemm<<<CB * CH, 256, (4 * BTILE_U32) * 4 + 1152 * 8 * 4 + 8 * 1024, stream>>>(
      x, (const unsigned*)b2, offset_buf, mask_buf, (float*)d_out);
}
